// HINormer_80865644249452
// MI455X (gfx1250) — compile-verified, matrix operand order checked by reading, not run on hardware
//
#include <hip/hip_runtime.h>
#include <hip/hip_bf16.h>

// ---------------------------------------------------------------------------
// HINormer-style GNN forward for MI455X (gfx1250, wave32, WMMA).
// - re_conv chain is dead code in the reference -> eliminated.
// - All GEMM operands staged in bf16 memory (N-major weights), so every WMMA
//   fragment is two contiguous 16B loads (global_load_b128) per lane.
// - GEMMs: v_wmma_f32_16x16x32_bf16 (f32 accumulate).
// - Edge aggregation: f32 messages + global_atomic_add_f32 (bandwidth stage).
// ---------------------------------------------------------------------------

typedef __attribute__((ext_vector_type(16))) __bf16 v16bf;
typedef __attribute__((ext_vector_type(8)))  float  v8f;

#define D_DIM  128   // hidden / input dim
#define SLEN_C 16    // sequence length (ego = seqs[:,0])
#define C_DIM  16    // classes
#define T_NUM  4     // node types
#define KB_CNT 4     // 128 / 32 K-chunks per WMMA accumulation chain

// Fragment load from bf16 memory laid out so that the 16 elements a lane
// needs are at p[0..7] and p[16..23] (each 16B-aligned, contiguous).
// Works for both A (row-major activations) and B (N-major weights).
__device__ __forceinline__ v16bf load_frag(const __bf16* __restrict__ p) {
  v16bf f;
#pragma unroll
  for (int e = 0; e < 8; ++e) {
    f[e]     = p[e];        // K = k0 + g*8 + e
    f[8 + e] = p[16 + e];   // K = k0 + 16 + g*8 + e
  }
  return f;
}

// --------------------------- prep / convert kernels ------------------------

// float4-wide f32 -> bf16 convert (n must be a multiple of 4).
__global__ void cvt_bf16_kernel(const float* __restrict__ src,
                                __bf16* __restrict__ dst, int n4) {
  int i = blockIdx.x * blockDim.x + threadIdx.x;
  if (i >= n4) return;
  const float4 v = ((const float4*)src)[i];
  __bf16* d = dst + (size_t)i * 4;
  d[0] = (__bf16)v.x; d[1] = (__bf16)v.y;
  d[2] = (__bf16)v.z; d[3] = (__bf16)v.w;
}

// Batched transpose+convert: src = batch of (K x N) f32, dst = batch of
// (N x K) bf16.  dst[b][n][k] = src[b][k][n]
__global__ void transpose_cvt_kernel(const float* __restrict__ src,
                                     __bf16* __restrict__ dst,
                                     int K, int N, int total) {
  int i = blockIdx.x * blockDim.x + threadIdx.x;
  if (i >= total) return;
  int kn = K * N;
  int b  = i / kn;
  int r  = i - b * kn;
  int k  = r / N;
  int n  = r - k * N;
  dst[(size_t)b * kn + (size_t)n * K + k] = (__bf16)src[i];
}

// --------------------------- degree / norm kernels -------------------------

__global__ void zero_int_kernel(int* __restrict__ p, int n) {
  int i = blockIdx.x * blockDim.x + threadIdx.x;
  if (i < n) p[i] = 0;
}

__global__ void count_col_kernel(const int* __restrict__ col,
                                 int* __restrict__ cnt, int E) {
  int e = blockIdx.x * blockDim.x + threadIdx.x;
  if (e < E) atomicAdd(&cnt[col[e]], 1);
}

__global__ void make_dis_kernel(const int* __restrict__ cnt,
                                float* __restrict__ dis, int N) {
  int n = blockIdx.x * blockDim.x + threadIdx.x;
  if (n < N) dis[n] = rsqrtf((float)(cnt[n] + 1));  // +1 self loop
}

// ------------------------------ typed FC (WMMA) ----------------------------
// ghb[n] = bf16( xb[n] @ fcWt[type[n]]^T + fcB[type[n]] )
// Block = 256 threads (8 waves); wave w covers output cols [w*16, w*16+16).
__global__ void fc_typed_kernel(const __bf16* __restrict__ xb,
                                const int* __restrict__ node_type,
                                const __bf16* __restrict__ fcWt,  // T x N x K
                                const float* __restrict__ fcB,
                                __bf16* __restrict__ ghb, int Nnodes) {
  const int wave = threadIdx.x >> 5;
  const int lane = threadIdx.x & 31;
  const int row0 = blockIdx.x * 16;
  const int n0   = wave * 16;
  const int m    = lane & 15;
  const int g    = lane >> 4;
  const int arow = min(row0 + m, Nnodes - 1);
  const int myType = node_type[arow];

  const __bf16* abase = xb + (size_t)arow * D_DIM + g * 8;

  // Types of the 8 rows this lane will store (M = g*8 + i).
  int tt[8];
#pragma unroll
  for (int i = 0; i < 8; ++i)
    tt[i] = node_type[min(row0 + g * 8 + i, Nnodes - 1)];

  float outv[8];
#pragma unroll
  for (int i = 0; i < 8; ++i) outv[i] = 0.0f;

  for (int t = 0; t < T_NUM; ++t) {
    if (__ballot(myType == t) == 0ull) continue;   // wave-uniform skip
    const __bf16* bbase = fcWt + (size_t)t * D_DIM * D_DIM
                        + (size_t)(n0 + m) * D_DIM + g * 8;
    v8f acc = {};
#pragma unroll
    for (int kb = 0; kb < KB_CNT; ++kb) {
      v16bf af = load_frag(abase + kb * 32);
      v16bf bf = load_frag(bbase + kb * 32);
      acc = __builtin_amdgcn_wmma_f32_16x16x32_bf16(
          false, af, false, bf, (short)0, acc, false, false);
    }
    const float bias = fcB[(size_t)t * D_DIM + n0 + m];
#pragma unroll
    for (int i = 0; i < 8; ++i)
      outv[i] = (tt[i] == t) ? (acc[i] + bias) : outv[i];   // select, no store
  }

  // Single unconditional store pass.
#pragma unroll
  for (int i = 0; i < 8; ++i) {
    const int r = row0 + g * 8 + i;
    if (r < Nnodes) ghb[(size_t)r * D_DIM + n0 + m] = (__bf16)outv[i];
  }
}

// --------------------------- dense 128x128 GEMM ---------------------------
// C[M x 128] f32 = A[M x 128] bf16 @ Bt[128 x 128]^T bf16
__global__ void gemm128_kernel(const __bf16* __restrict__ A,
                               const __bf16* __restrict__ Bt,  // N-major
                               float* __restrict__ C, int Nrows) {
  const int wave = threadIdx.x >> 5;
  const int lane = threadIdx.x & 31;
  const int row0 = blockIdx.x * 16;
  const int n0   = wave * 16;
  const int m    = lane & 15;
  const int g    = lane >> 4;
  const int arow = min(row0 + m, Nrows - 1);

  const __bf16* abase = A  + (size_t)arow * D_DIM + g * 8;
  const __bf16* bbase = Bt + (size_t)(n0 + m) * D_DIM + g * 8;

  v8f acc = {};
#pragma unroll
  for (int kb = 0; kb < KB_CNT; ++kb) {
    v16bf af = load_frag(abase + kb * 32);
    v16bf bf = load_frag(bbase + kb * 32);
    acc = __builtin_amdgcn_wmma_f32_16x16x32_bf16(
        false, af, false, bf, (short)0, acc, false, false);
  }
#pragma unroll
  for (int i = 0; i < 8; ++i) {
    const int r = row0 + g * 8 + i;
    if (r < Nrows) C[(size_t)r * D_DIM + n0 + m] = acc[i];
  }
}

// --------------------------- GCN aggregation -------------------------------
// acc[n][d] = gcnB[d] + dis[n]^2 * xw[n][d]   (bias + self-loop message)
// float4-wide: one thread handles 4 consecutive d.
__global__ void init_acc_kernel(const float* __restrict__ xw,
                                const float* __restrict__ dis,
                                const float* __restrict__ bias,
                                float* __restrict__ acc, int N) {
  int i = blockIdx.x * blockDim.x + threadIdx.x;          // over N * 32
  if (i >= N * (D_DIM / 4)) return;
  int n  = i >> 5;
  int d4 = (i & 31) * 4;
  float s  = dis[n];
  float s2 = s * s;
  const float4 v = *(const float4*)(xw + (size_t)n * D_DIM + d4);
  const float4 b = *(const float4*)(bias + d4);
  float4 o;
  o.x = b.x + s2 * v.x; o.y = b.y + s2 * v.y;
  o.z = b.z + s2 * v.z; o.w = b.w + s2 * v.w;
  *(float4*)(acc + (size_t)n * D_DIM + d4) = o;
}

// One wave per edge; lane handles 4 consecutive floats of the 128-wide msg.
__global__ void edge_scatter_kernel(const int* __restrict__ row,
                                    const int* __restrict__ col,
                                    const float* __restrict__ xw,
                                    const float* __restrict__ dis,
                                    float* __restrict__ acc, int E) {
  int t = blockIdx.x * blockDim.x + threadIdx.x;
  int e = t >> 5;
  if (e >= E) return;
  int lane = t & 31;
  int r = row[e];
  int c = col[e];
  float norm = dis[r] * dis[c];
  const float4 v = *(const float4*)(xw + (size_t)r * D_DIM + lane * 4);
  float* dst = acc + (size_t)c * D_DIM + lane * 4;
  atomicAdd(dst + 0, v.x * norm);
  atomicAdd(dst + 1, v.y * norm);
  atomicAdd(dst + 2, v.z * norm);
  atomicAdd(dst + 3, v.w * norm);
}

// ReLU + re-quantize to bf16 for the next GEMM stage (float4-wide).
__global__ void relu_bf16_kernel(const float* __restrict__ src,
                                 __bf16* __restrict__ dst, int n4) {
  int i = blockIdx.x * blockDim.x + threadIdx.x;
  if (i >= n4) return;
  const float4 v = ((const float4*)src)[i];
  __bf16* d = dst + (size_t)i * 4;
  d[0] = (__bf16)fmaxf(v.x, 0.0f);
  d[1] = (__bf16)fmaxf(v.y, 0.0f);
  d[2] = (__bf16)fmaxf(v.z, 0.0f);
  d[3] = (__bf16)fmaxf(v.w, 0.0f);
}

// ------------------------------- prediction --------------------------------
// out[b] = ghb[ego[b]] @ predWt^T + predB ; one wave per 16-ego tile.
__global__ void pred_kernel(const __bf16* __restrict__ ghb,
                            const int* __restrict__ seqs,
                            const __bf16* __restrict__ predWt, // C x K
                            const float* __restrict__ predB,
                            float* __restrict__ out, int Bseq) {
  const int lane = threadIdx.x & 31;
  const int row0 = blockIdx.x * 16;
  const int m    = lane & 15;
  const int g    = lane >> 4;
  const int brow = min(row0 + m, Bseq - 1);
  const int arow = seqs[(size_t)brow * SLEN_C];   // ego = seqs[:,0]

  const __bf16* abase = ghb + (size_t)arow * D_DIM + g * 8;
  const __bf16* bbase = predWt + (size_t)m * D_DIM + g * 8;

  v8f acc = {};
#pragma unroll
  for (int kb = 0; kb < KB_CNT; ++kb) {
    v16bf af = load_frag(abase + kb * 32);
    v16bf bf = load_frag(bbase + kb * 32);
    acc = __builtin_amdgcn_wmma_f32_16x16x32_bf16(
        false, af, false, bf, (short)0, acc, false, false);
  }
  const float bias = predB[m];
#pragma unroll
  for (int i = 0; i < 8; ++i) {
    const int r = row0 + g * 8 + i;
    if (r < Bseq) out[(size_t)r * C_DIM + m] = acc[i] + bias;
  }
}

__global__ void label_kernel(const int* __restrict__ label,
                             const int* __restrict__ seqs,
                             float* __restrict__ out, int Bseq) {
  int i = blockIdx.x * blockDim.x + threadIdx.x;
  if (i < Bseq) out[i] = (float)label[seqs[(size_t)i * SLEN_C]];
}

// ------------------------------- launcher ----------------------------------

extern "C" void kernel_launch(void* const* d_in, const int* in_sizes, int n_in,
                              void* d_out, int out_size, void* d_ws, size_t ws_size,
                              hipStream_t stream) {
  const float* x         = (const float*)d_in[0];
  const int*   label     = (const int*)d_in[1];
  const int*   seqs      = (const int*)d_in[2];
  const int*   edge      = (const int*)d_in[3];
  const int*   node_type = (const int*)d_in[4];
  const float* fcW       = (const float*)d_in[5];
  const float* fcB       = (const float*)d_in[6];
  const float* gcnW      = (const float*)d_in[7];
  const float* gcnB      = (const float*)d_in[8];
  // d_in[9..11] (reW / re_wtype / reB) feed a value never used by the output.
  const float* predW     = (const float*)d_in[12];
  const float* predB     = (const float*)d_in[13];

  const int N    = in_sizes[4];
  const int E    = in_sizes[3] / 2;
  const int Bseq = in_sizes[2] / SLEN_C;
  const int L    = in_sizes[8] / D_DIM;   // gcnB is (L, D)

  const int* erow = edge;       // edge_index[0]
  const int* ecol = edge + E;   // edge_index[1]

  // Workspace carve-up (256B aligned).
  char*  ws  = (char*)d_ws;
  size_t off = 0;
  auto carve = [&](size_t bytes) {
    void* p = ws + off;
    off = (off + bytes + 255) & ~(size_t)255;
    return p;
  };
  __bf16* ghb    = (__bf16*)carve((size_t)N * D_DIM * sizeof(__bf16));
  __bf16* xb     = (__bf16*)carve((size_t)N * D_DIM * sizeof(__bf16));
  float*  xw     = (float*)carve((size_t)N * D_DIM * sizeof(float));
  float*  acc    = (float*)carve((size_t)N * D_DIM * sizeof(float));
  float*  dis    = (float*)carve((size_t)N * sizeof(float));
  int*    cnt    = (int*)carve((size_t)N * sizeof(int));
  __bf16* fcWt   = (__bf16*)carve((size_t)T_NUM * D_DIM * D_DIM * sizeof(__bf16));
  __bf16* gcnWt  = (__bf16*)carve((size_t)L * D_DIM * D_DIM * sizeof(__bf16));
  __bf16* predWt = (__bf16*)carve((size_t)C_DIM * D_DIM * sizeof(__bf16));
  (void)ws_size; (void)n_in; (void)out_size;

  const int mtiles = (N + 15) / 16;
  const int nd     = N * D_DIM;
  const int nd4    = nd / 4;

  // 0) one-time bf16 staging: x copy + weights transposed to N-major bf16.
  cvt_bf16_kernel<<<(nd4 + 255) / 256, 256, 0, stream>>>(x, xb, nd4);
  {
    int tot = T_NUM * D_DIM * D_DIM;
    transpose_cvt_kernel<<<(tot + 255) / 256, 256, 0, stream>>>(
        fcW, fcWt, D_DIM, D_DIM, tot);
  }
  {
    int tot = L * D_DIM * D_DIM;
    transpose_cvt_kernel<<<(tot + 255) / 256, 256, 0, stream>>>(
        gcnW, gcnWt, D_DIM, D_DIM, tot);
  }
  {
    int tot = D_DIM * C_DIM;
    transpose_cvt_kernel<<<(tot + 255) / 256, 256, 0, stream>>>(
        predW, predWt, D_DIM, C_DIM, tot);
  }

  // 1) degree -> dis = (deg_col + 1)^-1/2
  zero_int_kernel<<<(N + 255) / 256, 256, 0, stream>>>(cnt, N);
  count_col_kernel<<<(E + 255) / 256, 256, 0, stream>>>(ecol, cnt, E);
  make_dis_kernel<<<(N + 255) / 256, 256, 0, stream>>>(cnt, dis, N);

  // 2) typed FC -> ghb  (WMMA bf16)
  fc_typed_kernel<<<mtiles, 256, 0, stream>>>(xb, node_type, fcWt, fcB, ghb, N);

  // 3) GCN layers
  for (int l = 0; l < L; ++l) {
    gemm128_kernel<<<mtiles, 256, 0, stream>>>(
        ghb, gcnWt + (size_t)l * D_DIM * D_DIM, xw, N);
    init_acc_kernel<<<(N * 32 + 255) / 256, 256, 0, stream>>>(
        xw, dis, gcnB + (size_t)l * D_DIM, acc, N);
    edge_scatter_kernel<<<((size_t)E * 32 + 255) / 256, 256, 0, stream>>>(
        erow, ecol, xw, dis, acc, E);
    relu_bf16_kernel<<<(nd4 + 255) / 256, 256, 0, stream>>>(acc, ghb, nd4);
  }

  // 4) prediction + labels
  pred_kernel<<<(Bseq + 15) / 16, 32, 0, stream>>>(
      ghb, seqs, predWt, predB, (float*)d_out, Bseq);
  label_kernel<<<(Bseq + 255) / 256, 256, 0, stream>>>(
      label, seqs, (float*)d_out + (size_t)Bseq * C_DIM, Bseq);
}
